// SimpleMLP_LoRA_58978490908683
// MI455X (gfx1250) — matrix operand order repaired and checked by reading, not compiled, not run on hardware
//
#include <hip/hip_runtime.h>

#define BATCH_N 524288
#define DIN 128
#define DHID 512
#define DOUT 128
#define RANK 16

typedef __attribute__((ext_vector_type(16))) __bf16 v16bf;
typedef __attribute__((ext_vector_type(8)))  float  v8f;

union Frag {
    v16bf v;
    unsigned d[8];
    uint4 q[2];
};

// Pack two f32 -> one dword of two bf16 (round-to-nearest via +0x8000, then
// grab the two high halves with a single v_perm_b32).
__device__ __forceinline__ unsigned packbf(float a, float b) {
    unsigned ua = __builtin_bit_cast(unsigned, a) + 0x8000u;
    unsigned ub = __builtin_bit_cast(unsigned, b) + 0x8000u;
    return __builtin_amdgcn_perm(ub, ua, 0x07060302u);  // {ub.hi16, ua.hi16}
}

__device__ __forceinline__ unsigned short f2bf(float f) {
    unsigned u = __builtin_bit_cast(unsigned, f) + 0x8000u;
    return (unsigned short)(u >> 16);
}

// ---------------------------------------------------------------------------
// Prep: W1 = A1@B1 -> bf16 [DHID x DIN]; W2 = A2@B2 -> bf16 [DOUT x DHID].
// ---------------------------------------------------------------------------
__global__ void lora_prep(const float* __restrict__ A1, const float* __restrict__ B1,
                          const float* __restrict__ A2, const float* __restrict__ B2,
                          unsigned short* __restrict__ w) {
    int idx = blockIdx.x * blockDim.x + threadIdx.x;
    if (idx < DHID * DIN) {
        int hd = idx / DIN, i = idx % DIN;
        float s = 0.f;
        #pragma unroll
        for (int r = 0; r < RANK; ++r) s += A1[hd * RANK + r] * B1[r * DIN + i];
        w[idx] = f2bf(s);
    } else {
        int j = idx - DHID * DIN;
        if (j < DOUT * DHID) {
            int o = j / DHID, hd = j % DHID;
            float s = 0.f;
            #pragma unroll
            for (int r = 0; r < RANK; ++r) s += A2[o * RANK + r] * B2[r * DHID + hd];
            w[DHID * DIN + j] = f2bf(s);
        }
    }
}

// ---------------------------------------------------------------------------
// Fused transposed-GEMM kernel:
//   h^T_tile = W1_tile(A) x x^T(B);  square;  y^T_tile = W2_tile(A) x h^T(B)
// Weights live in LDS (256 KB); h transpose = one XOR-16 half-wave shuffle.
// ---------------------------------------------------------------------------
__global__ __launch_bounds__(256, 1)
void lora_main(const float* __restrict__ x, const unsigned short* __restrict__ w,
               float* __restrict__ y) {
    __shared__ unsigned short sW1[DHID * DIN];     // 128 KB  bf16 W1 [hid][in]
    __shared__ unsigned short sW2[DOUT * DHID];    // 128 KB  bf16 W2 [out][hid]

    const int tid  = threadIdx.x;
    const int wave = tid >> 5;
    const int lane = tid & 31;
    const int m    = lane & 15;    // A-row / B-column / batch-column index
    const int half = lane >> 4;    // half-wave select

    // Cooperative weight fill (uint4 = 8 bf16 per transfer).
    {
        const uint4* gsrc = (const uint4*)w;
        uint4* dst1 = (uint4*)sW1;
        uint4* dst2 = (uint4*)sW2;
        const int n16 = (DHID * DIN) / 8;          // 8192 uint4 per matrix
        for (int i = tid; i < n16; i += 256) dst1[i] = gsrc[i];
        for (int i = tid; i < n16; i += 256) dst2[i] = gsrc[n16 + i];
    }
    __syncthreads();

    const int nTasks  = BATCH_N / 32;              // 16384 tasks of 32 rows
    const int wstride = gridDim.x * 8;

    for (int task = blockIdx.x * 8 + wave; task < nTasks; task += wstride) {
        const int rowBase = task * 32;

        // --- x^T B-fragments xb[g][c]: lane = batch column (m), elements =
        // 16 consecutive K (input dim) starting at c*32 + 16*half.
        Frag xb[2][4];
        #pragma unroll
        for (int g = 0; g < 2; ++g) {
            const float* xr = x + (size_t)(rowBase + g * 16 + m) * DIN;
            #pragma unroll
            for (int c = 0; c < 4; ++c) {
                const float* p = xr + c * 32 + half * 16;
                float4 p0 = *(const float4*)(p);
                float4 p1 = *(const float4*)(p + 4);
                float4 p2 = *(const float4*)(p + 8);
                float4 p3 = *(const float4*)(p + 12);
                xb[g][c].d[0] = packbf(p0.x, p0.y);
                xb[g][c].d[1] = packbf(p0.z, p0.w);
                xb[g][c].d[2] = packbf(p1.x, p1.y);
                xb[g][c].d[3] = packbf(p1.z, p1.w);
                xb[g][c].d[4] = packbf(p2.x, p2.y);
                xb[g][c].d[5] = packbf(p2.z, p2.w);
                xb[g][c].d[6] = packbf(p3.x, p3.y);
                xb[g][c].d[7] = packbf(p3.z, p3.w);
            }
        }

        v8f acc[2][8] = {};                        // y^T accumulators [g][out-tile]

        #pragma unroll 1
        for (int jj = 0; jj < DHID / 32; ++jj) {   // K=32 slab of the hidden dim
            // --- GEMM1: h^T tiles. A = W1 rows (hid), B = x^T.
            v8f hT[2][2] = {};                     // [g][t], t = hid sub-tile
            #pragma unroll
            for (int c = 0; c < 4; ++c) {
                // A layout (16-bit 16x32): lane m = row, elements = K at
                // {c*32 + half*16 .. +16} (contiguous, matching xb).
                Frag a0, a1;
                const unsigned short* q0 = &sW1[((2 * jj)     * 16 + m) * DIN + c * 32 + half * 16];
                const unsigned short* q1 = &sW1[((2 * jj + 1) * 16 + m) * DIN + c * 32 + half * 16];
                a0.q[0] = *(const uint4*)q0;  a0.q[1] = *(const uint4*)(q0 + 8);
                a1.q[0] = *(const uint4*)q1;  a1.q[1] = *(const uint4*)(q1 + 8);
                #pragma unroll
                for (int g = 0; g < 2; ++g) {
                    hT[g][0] = __builtin_amdgcn_wmma_f32_16x16x32_bf16(
                        false, a0.v, false, xb[g][c].v, (short)0, hT[g][0], false, false);
                    hT[g][1] = __builtin_amdgcn_wmma_f32_16x16x32_bf16(
                        false, a1.v, false, xb[g][c].v, (short)0, hT[g][1], false, false);
                }
            }

            // --- Square + build h^T B-fragments. C/D gives lane=batch,
            // VGPR r = hid_local r + 8*half. B-frag needs lane=batch,
            // elements = hid_local 0..15 of tile 'half'. So each lane keeps
            // its 8 values of its own tile and swaps the other tile's 8
            // values with lane^16 (packed as 4 dwords of bf16 pairs).
            Frag hb[2];
            #pragma unroll
            for (int g = 0; g < 2; ++g) {
                unsigned own0[4], own1[4];
                #pragma unroll
                for (int r2 = 0; r2 < 4; ++r2) {
                    float a0s = hT[g][0][2 * r2],     a1s = hT[g][0][2 * r2 + 1];
                    float b0s = hT[g][1][2 * r2],     b1s = hT[g][1][2 * r2 + 1];
                    own0[r2] = packbf(a0s * a0s, a1s * a1s);
                    own1[r2] = packbf(b0s * b0s, b1s * b1s);
                }
                #pragma unroll
                for (int r2 = 0; r2 < 4; ++r2) {
                    unsigned snd = half ? own0[r2] : own1[r2];
                    unsigned rcv = (unsigned)__shfl_xor((int)snd, 16, 32);
                    hb[g].d[r2]     = half ? rcv      : own0[r2];
                    hb[g].d[4 + r2] = half ? own1[r2] : rcv;
                }
            }

            // --- GEMM2: y^T accumulate. A = W2 rows (out), B = h^T.
            #pragma unroll
            for (int nt = 0; nt < 8; ++nt) {
                Frag a2;
                const unsigned short* q = &sW2[(nt * 16 + m) * DHID + jj * 32 + half * 16];
                a2.q[0] = *(const uint4*)q;  a2.q[1] = *(const uint4*)(q + 8);
                #pragma unroll
                for (int g = 0; g < 2; ++g)
                    acc[g][nt] = __builtin_amdgcn_wmma_f32_16x16x32_bf16(
                        false, a2.v, false, hb[g].v, (short)0, acc[g][nt], false, false);
            }
        }

        // --- Store y. y^T C/D layout: lane = batch (m), VGPR r -> out =
        // nt*16 + 8*half + r => 8 contiguous floats per lane per tile.
        #pragma unroll
        for (int g = 0; g < 2; ++g) {
            float* yr = y + (size_t)(rowBase + g * 16 + m) * DOUT + half * 8;
            #pragma unroll
            for (int nt = 0; nt < 8; ++nt) {
                float4 lo = { acc[g][nt][0], acc[g][nt][1], acc[g][nt][2], acc[g][nt][3] };
                float4 hi = { acc[g][nt][4], acc[g][nt][5], acc[g][nt][6], acc[g][nt][7] };
                *(float4*)(yr + nt * 16)     = lo;
                *(float4*)(yr + nt * 16 + 4) = hi;
            }
        }
    }
}

extern "C" void kernel_launch(void* const* d_in, const int* in_sizes, int n_in,
                              void* d_out, int out_size, void* d_ws, size_t ws_size,
                              hipStream_t stream) {
    const float* x  = (const float*)d_in[0];
    const float* A1 = (const float*)d_in[1];
    const float* B1 = (const float*)d_in[2];
    const float* A2 = (const float*)d_in[3];
    const float* B2 = (const float*)d_in[4];
    float* out = (float*)d_out;
    unsigned short* wbf = (unsigned short*)d_ws;   // 256 KB of workspace used

    lora_prep<<<(DHID * DIN + DOUT * DHID + 255) / 256, 256, 0, stream>>>(A1, B1, A2, B2, wbf);
    lora_main<<<1024, 256, 0, stream>>>(x, wbf, out);
}
